// MultiHeadAttention_6416681140585
// MI455X (gfx1250) — compile-verified, matrix operand order checked by reading, not run on hardware
//
#include <hip/hip_runtime.h>
#include <hip/hip_bf16.h>

// ---------------------------------------------------------------------------
// MHA forward for MI455X (gfx1250, wave32, WMMA).
//   B=8, K=1024, D_MODEL=1024, H=16, HD=64.  M = B*K = 8192.
// Pipeline:
//   cvt f32->bf16 (q,k,v) ; transpose+cvt weights -> Wt[N][K] bf16
//   GEMM bf16 (WMMA, 64x64/wave) : Qp/Kp/Vp projections (bf16 out, +bias)
//   Flash attention (WMMA)       : ctx bf16 (scores never materialized)
//   GEMM bf16 (WMMA, 64x64/wave) : ctx @ WoT + bo -> f32 d_out
// ---------------------------------------------------------------------------

#define BATCH   8
#define SEQ     1024
#define DMODEL  1024
#define NHEAD   16
#define HDIM    64
#define MROWS   (BATCH * SEQ)          // 8192

typedef __attribute__((ext_vector_type(16))) __bf16 bf16x16;
typedef __attribute__((ext_vector_type(8)))  float  f32x8;

union FragU { bf16x16 v; uint4 q[2]; };
union V8U   { uint4 q; __bf16 e[8]; };

// Load one 16-element bf16 WMMA fragment for this lane.
// Caller passes p already offset by row/col and the 8*(lane>>4) K-base.
// Covers K = [0..7] and [16..23] relative to p (ISA 16-bit A/B layout).
__device__ __forceinline__ bf16x16 ldfrag(const __bf16* p) {
    FragU u;
    u.q[0] = *reinterpret_cast<const uint4*>(p);
    u.q[1] = *reinterpret_cast<const uint4*>(p + 16);
    return u.v;
}

__device__ __forceinline__ f32x8 wmma_bf16(bf16x16 a, bf16x16 b, f32x8 c) {
    return __builtin_amdgcn_wmma_f32_16x16x32_bf16(
        /*neg_a=*/false, a, /*neg_b=*/false, b,
        /*c_mod=*/(short)0, c, /*reuse_a=*/false, /*reuse_b=*/false);
}

// ---------------------------------------------------------------------------
// f32 -> bf16 elementwise (n divisible by 4)
// ---------------------------------------------------------------------------
__global__ void cvt_bf16_kernel(const float* __restrict__ in,
                                __bf16* __restrict__ out, int n) {
    int i = (blockIdx.x * blockDim.x + threadIdx.x) * 4;
    if (i + 3 < n) {
        float4 f = *reinterpret_cast<const float4*>(in + i);
        out[i + 0] = (__bf16)f.x;
        out[i + 1] = (__bf16)f.y;
        out[i + 2] = (__bf16)f.z;
        out[i + 3] = (__bf16)f.w;
    }
}

// ---------------------------------------------------------------------------
// W [D][N] f32  ->  Wt [N][D] bf16   (32x32 LDS tile transpose)
// block (32,8), grid (N/32, D/32)
// ---------------------------------------------------------------------------
__global__ void transpose_bf16_kernel(const float* __restrict__ in,
                                      __bf16* __restrict__ out,
                                      int D, int N) {
    __shared__ float tile[32][33];
    int n0 = blockIdx.x * 32, d0 = blockIdx.y * 32;
    #pragma unroll
    for (int i = 0; i < 32; i += 8)
        tile[threadIdx.y + i][threadIdx.x] =
            in[(size_t)(d0 + threadIdx.y + i) * N + n0 + threadIdx.x];
    __syncthreads();
    #pragma unroll
    for (int i = 0; i < 32; i += 8)
        out[(size_t)(n0 + threadIdx.y + i) * D + d0 + threadIdx.x] =
            (__bf16)tile[threadIdx.x][threadIdx.y + i];
}

// ---------------------------------------------------------------------------
// C[M,N] = A[M,Kd] * Bt[N,Kd]^T + bias      (bf16 in, f32 acc)
// Each wave owns a 64x64 output tile: per K=32 step, 8 fragment loads feed
// 16 WMMAs (32 FLOP/byte from L2; operands are L2-resident).
// block (32,8) = 8 waves; grid.x = (M/64)*(N/64)/8.
// outb != nullptr -> bf16 out; else f32 to outf.
// ---------------------------------------------------------------------------
__global__ __launch_bounds__(256)
void gemm_bf16_kernel(const __bf16* __restrict__ A,
                      const __bf16* __restrict__ Bt,
                      const float*  __restrict__ bias,
                      __bf16* __restrict__ outb,
                      float*  __restrict__ outf,
                      int M, int N, int Kd) {
    const int lane = threadIdx.x;          // 0..31
    const int wid  = threadIdx.y;          // 0..7
    const int hi   = lane >> 4;            // lane half
    const int l16  = lane & 15;
    const int tilesN = N / 64;
    int id = blockIdx.x * 8 + wid;
    int tm = (id / tilesN) * 64;
    int tn = (id % tilesN) * 64;
    if (tm >= M) return;

    const __bf16* aP[4];
    const __bf16* bP[4];
    #pragma unroll
    for (int i = 0; i < 4; ++i) {
        aP[i] = A  + (size_t)(tm + i * 16 + l16) * Kd + 8 * hi;
        bP[i] = Bt + (size_t)(tn + i * 16 + l16) * Kd + 8 * hi;
    }

    f32x8 c[4][4] = {};
    for (int k = 0; k < Kd; k += 32) {
        bf16x16 fa[4], fb[4];
        #pragma unroll
        for (int i = 0; i < 4; ++i) fa[i] = ldfrag(aP[i] + k);
        #pragma unroll
        for (int i = 0; i < 4; ++i) fb[i] = ldfrag(bP[i] + k);
        #pragma unroll
        for (int i = 0; i < 4; ++i)
            #pragma unroll
            for (int j = 0; j < 4; ++j)
                c[i][j] = wmma_bf16(fa[i], fb[j], c[i][j]);
    }

    #pragma unroll
    for (int j = 0; j < 4; ++j) {
        int col = tn + j * 16 + l16;
        float bb = bias[col];
        #pragma unroll
        for (int i = 0; i < 4; ++i) {
            #pragma unroll
            for (int r = 0; r < 8; ++r) {
                size_t idx = (size_t)(tm + i * 16 + r + 8 * hi) * N + col;
                float v = c[i][j][r] + bb;
                if (outf) outf[idx] = v;
                else      outb[idx] = (__bf16)v;
            }
        }
    }
}

// ---------------------------------------------------------------------------
// Flash attention.  block = (32,8): 8 waves x 16 query rows = 128 rows.
// grid = (SEQ/128, NHEAD, BATCH).  32 keys per tile, online softmax.
// Qp/Kp/Vp layout: [B*SEQ][NHEAD*HDIM] bf16. ctx same layout, bf16.
// Scale = 1/sqrt(SEQ) = 1/32 (matches reference).
// ---------------------------------------------------------------------------
__global__ __launch_bounds__(256)
void flash_attn_kernel(const __bf16* __restrict__ Qp,
                       const __bf16* __restrict__ Kp,
                       const __bf16* __restrict__ Vp,
                       __bf16* __restrict__ ctx) {
    __shared__ __align__(16) __bf16 kt[32 * 64];       // K-tile, row-major [key][d]
    __shared__ __align__(16) __bf16 vt[64 * 40];       // V-tile transposed [d][key], pad 40
    __shared__ __align__(16) __bf16 pt[8][16 * 32];    // per-wave P transpose scratch

    const int lane = threadIdx.x, wid = threadIdx.y;
    const int hi = lane >> 4, l16 = lane & 15;
    const int tid = wid * 32 + lane;
    const int b = blockIdx.z, h = blockIdx.y;
    const int rowBase = blockIdx.x * 128 + wid * 16;

    const size_t hOff = (size_t)h * HDIM;
    const size_t qRow = ((size_t)b * SEQ + rowBase);

    // Resident Q fragments: A layout, row = l16, two d-chunks (0..31, 32..63)
    const __bf16* qptr = Qp + (qRow + l16) * (NHEAD * HDIM) + hOff + 8 * hi;
    bf16x16 qf0 = ldfrag(qptr);
    bf16x16 qf1 = ldfrag(qptr + 32);

    f32x8 o0 = {}, o1 = {}, o2 = {}, o3 = {};
    float mrow[8], lrow[8];
    #pragma unroll
    for (int r = 0; r < 8; ++r) { mrow[r] = -1.0e30f; lrow[r] = 0.0f; }

    const float scale = 0.03125f;   // 1/sqrt(1024)

    for (int j0 = 0; j0 < SEQ; j0 += 32) {
        __syncthreads();
        // --- cooperative stage: K tile (row-major), b128 coalesced -------
        {
            int row = tid >> 3, ch = tid & 7;                 // 32 rows x 8 chunks
            const __bf16* src = Kp + ((size_t)b * SEQ + j0 + row) * (NHEAD * HDIM)
                                   + hOff + ch * 8;
            *reinterpret_cast<uint4*>(&kt[row * 64 + ch * 8]) =
                *reinterpret_cast<const uint4*>(src);
        }
        // --- cooperative stage: V tile transposed [d][key] ---------------
        // One coalesced b128 per thread (8 d-contiguous bf16 of one key),
        // then 8 ds_store_b16 into the transposed tile.
        {
            int key = tid & 31, dch = tid >> 5;               // 32 keys x 8 d-chunks
            V8U u;
            u.q = *reinterpret_cast<const uint4*>(
                Vp + ((size_t)b * SEQ + j0 + key) * (NHEAD * HDIM) + hOff + dch * 8);
            #pragma unroll
            for (int t = 0; t < 8; ++t)
                vt[(dch * 8 + t) * 40 + key] = u.e[t];
        }
        __syncthreads();

        // --- scores: S[16 rows x 32 keys] --------------------------------
        bf16x16 kf0  = ldfrag(&kt[l16 * 64 + 8 * hi]);
        bf16x16 kf0b = ldfrag(&kt[l16 * 64 + 32 + 8 * hi]);
        bf16x16 kf1  = ldfrag(&kt[(16 + l16) * 64 + 8 * hi]);
        bf16x16 kf1b = ldfrag(&kt[(16 + l16) * 64 + 32 + 8 * hi]);
        f32x8 s0 = {}, s1 = {};
        s0 = wmma_bf16(qf0, kf0,  s0);
        s0 = wmma_bf16(qf1, kf0b, s0);
        s1 = wmma_bf16(qf0, kf1,  s1);
        s1 = wmma_bf16(qf1, kf1b, s1);

        // --- online softmax (row stats across 16 lanes of each half) -----
        #pragma unroll
        for (int r = 0; r < 8; ++r) {
            float a = s0[r] * scale, c = s1[r] * scale;
            float mx = fmaxf(a, c);
            #pragma unroll
            for (int m = 1; m < 16; m <<= 1)
                mx = fmaxf(mx, __shfl_xor(mx, m, 32));
            float newM = fmaxf(mrow[r], mx);
            float corr = __expf(mrow[r] - newM);
            float p0 = __expf(a - newM);
            float p1 = __expf(c - newM);
            float rs = p0 + p1;
            #pragma unroll
            for (int m = 1; m < 16; m <<= 1)
                rs += __shfl_xor(rs, m, 32);
            lrow[r] = lrow[r] * corr + rs;
            mrow[r] = newM;
            o0[r] *= corr; o1[r] *= corr; o2[r] *= corr; o3[r] *= corr;
            int mrw = r + 8 * hi;                 // C-layout row of this element
            pt[wid][mrw * 32 + l16]      = (__bf16)p0;
            pt[wid][mrw * 32 + 16 + l16] = (__bf16)p1;
        }
        // wave-local LDS RAW: writes above feed reads below (in-order LDS)
        asm volatile("s_wait_dscnt 0" ::: "memory");

        // --- P (A layout) x V (B layout from vt) -------------------------
        bf16x16 pf = ldfrag(&pt[wid][l16 * 32 + 8 * hi]);
        o0 = wmma_bf16(pf, ldfrag(&vt[(0  + l16) * 40 + 8 * hi]), o0);
        o1 = wmma_bf16(pf, ldfrag(&vt[(16 + l16) * 40 + 8 * hi]), o1);
        o2 = wmma_bf16(pf, ldfrag(&vt[(32 + l16) * 40 + 8 * hi]), o2);
        o3 = wmma_bf16(pf, ldfrag(&vt[(48 + l16) * 40 + 8 * hi]), o3);
    }

    // --- epilogue: normalize and store ctx bf16 --------------------------
    #pragma unroll
    for (int r = 0; r < 8; ++r) {
        float inv = 1.0f / lrow[r];
        int m = r + 8 * hi;
        size_t ro = (qRow + m) * (NHEAD * HDIM) + hOff;
        ctx[ro + 0  + l16] = (__bf16)(o0[r] * inv);
        ctx[ro + 16 + l16] = (__bf16)(o1[r] * inv);
        ctx[ro + 32 + l16] = (__bf16)(o2[r] * inv);
        ctx[ro + 48 + l16] = (__bf16)(o3[r] * inv);
    }
}

// ---------------------------------------------------------------------------
// Launch
// ---------------------------------------------------------------------------
extern "C" void kernel_launch(void* const* d_in, const int* in_sizes, int n_in,
                              void* d_out, int out_size, void* d_ws, size_t ws_size,
                              hipStream_t stream) {
    (void)in_sizes; (void)n_in; (void)out_size; (void)ws_size;
    const float* q_in = (const float*)d_in[0];
    const float* k_in = (const float*)d_in[1];
    const float* v_in = (const float*)d_in[2];
    const float* Wq = (const float*)d_in[3];  const float* bq = (const float*)d_in[4];
    const float* Wk = (const float*)d_in[5];  const float* bk = (const float*)d_in[6];
    const float* Wv = (const float*)d_in[7];  const float* bv = (const float*)d_in[8];
    const float* Wo = (const float*)d_in[9];  const float* bo = (const float*)d_in[10];
    float* out = (float*)d_out;

    const size_t MK  = (size_t)MROWS * DMODEL;           // 8M elements
    const size_t MKB = MK * sizeof(__bf16);              // 16 MiB
    const size_t WB  = (size_t)DMODEL * DMODEL * sizeof(__bf16); // 2 MiB
    char* ws = (char*)d_ws;
    __bf16* xq  = (__bf16*)(ws);                 // act bf16 (aliased as ctx later)
    __bf16* xk  = (__bf16*)(ws + 1 * MKB);
    __bf16* xv  = (__bf16*)(ws + 2 * MKB);
    __bf16* wqT = (__bf16*)(ws + 3 * MKB);
    __bf16* wkT = (__bf16*)(ws + 3 * MKB + 1 * WB);
    __bf16* wvT = (__bf16*)(ws + 3 * MKB + 2 * WB);
    __bf16* woT = (__bf16*)(ws + 3 * MKB + 3 * WB);
    __bf16* qp  = (__bf16*)(ws + 3 * MKB + 4 * WB);
    __bf16* kp  = (__bf16*)(ws + 4 * MKB + 4 * WB);
    __bf16* vp  = (__bf16*)(ws + 5 * MKB + 4 * WB);
    __bf16* ctx = xq;   // xq dead after Q projection; reuse for context

    // activations -> bf16
    int cvtBlocks = (int)(MK / (256 * 4));
    cvt_bf16_kernel<<<cvtBlocks, 256, 0, stream>>>(q_in, xq, (int)MK);
    cvt_bf16_kernel<<<cvtBlocks, 256, 0, stream>>>(k_in, xk, (int)MK);
    cvt_bf16_kernel<<<cvtBlocks, 256, 0, stream>>>(v_in, xv, (int)MK);

    // weights -> bf16, transposed to [N][K]
    dim3 tb(32, 8), tg(DMODEL / 32, DMODEL / 32);
    transpose_bf16_kernel<<<tg, tb, 0, stream>>>(Wq, wqT, DMODEL, DMODEL);
    transpose_bf16_kernel<<<tg, tb, 0, stream>>>(Wk, wkT, DMODEL, DMODEL);
    transpose_bf16_kernel<<<tg, tb, 0, stream>>>(Wv, wvT, DMODEL, DMODEL);
    transpose_bf16_kernel<<<tg, tb, 0, stream>>>(Wo, woT, DMODEL, DMODEL);

    // projections (bf16 out), 64x64 tile per wave
    dim3 gb(32, 8);
    int gemmBlocks = (MROWS / 64) * (DMODEL / 64) / 8;   // 256
    gemm_bf16_kernel<<<gemmBlocks, gb, 0, stream>>>(xq, wqT, bq, qp, nullptr,
                                                    MROWS, DMODEL, DMODEL);
    gemm_bf16_kernel<<<gemmBlocks, gb, 0, stream>>>(xk, wkT, bk, kp, nullptr,
                                                    MROWS, DMODEL, DMODEL);
    gemm_bf16_kernel<<<gemmBlocks, gb, 0, stream>>>(xv, wvT, bv, vp, nullptr,
                                                    MROWS, DMODEL, DMODEL);

    // fused flash attention -> ctx (bf16)
    dim3 ag(SEQ / 128, NHEAD, BATCH);
    flash_attn_kernel<<<ag, gb, 0, stream>>>(qp, kp, vp, ctx);

    // output projection (f32 out + bias)
    gemm_bf16_kernel<<<gemmBlocks, gb, 0, stream>>>(ctx, woT, bo, nullptr, out,
                                                    MROWS, DMODEL, DMODEL);
}
